// CausalSelfAttention_66451734004011
// MI455X (gfx1250) — matrix operand order repaired and checked by reading, not compiled, not run on hardware
//
#include <hip/hip_runtime.h>
#include <hip/hip_bf16.h>

#define Bb 2
#define Tt 2048
#define Cc 2048
#define Hh 16
#define Dd 128

typedef __attribute__((ext_vector_type(16))) __bf16 v16bf;
typedef __attribute__((ext_vector_type(8)))  float  v8f;

__device__ __forceinline__ unsigned short f2bf(float f) {
  unsigned int u = __builtin_bit_cast(unsigned int, f);
  u += 0x7FFFu + ((u >> 16) & 1u);   // round-to-nearest-even
  return (unsigned short)(u >> 16);
}

// Load a 32-byte (16 x bf16) fragment from two 16B-aligned chunks.
__device__ __forceinline__ v16bf load2(const unsigned short* p0, const unsigned short* p1) {
  union { v16bf v; uint4 u[2]; } f;
  f.u[0] = *(const uint4*)p0;
  f.u[1] = *(const uint4*)p1;
  return f.v;
}

__device__ __forceinline__ v8f wmma_bf16(v16bf a, v16bf b, v8f c) {
  return __builtin_amdgcn_wmma_f32_16x16x32_bf16(false, a, false, b, (short)0, c,
                                                 false, false);
}

// Async copy 16B from global to LDS (per lane), tracked by ASYNCcnt.
__device__ __forceinline__ void async_copy_b128(const unsigned short* gsrc,
                                                unsigned short* lds_dst) {
  unsigned lds_off = (unsigned)(size_t)lds_dst;           // low 32b = LDS offset
  asm volatile("global_load_async_to_lds_b128 %0, %1, off"
               :: "v"(lds_off), "v"(gsrc)
               : "memory");
}

__device__ __forceinline__ void wait_async0() {
  asm volatile("s_wait_asynccnt 0x0" ::: "memory");
}

// ---------------------------------------------------------------------------
// fp32 -> bf16 elementwise convert
// ---------------------------------------------------------------------------
__global__ __launch_bounds__(256) void cvt_bf16(const float* __restrict__ in,
                                                unsigned short* __restrict__ out,
                                                int n) {
  int i = blockIdx.x * 256 + threadIdx.x;
  if (i < n) out[i] = f2bf(in[i]);
}

// ---------------------------------------------------------------------------
// C(MxN, f32) = A(MxK, bf16, row-major) * B(NxK, bf16, row-major)^T
// 128x128 macro tile, BK=32, 256 threads = 8 waves (2x4), wave tile 64x32.
// Double-buffered LDS filled by GLOBAL_LOAD_ASYNC_TO_LDS_B128; one barrier
// per K-step. LDS rows padded to 40 bf16 (80B) for conflict-free b128 gathers.
// ---------------------------------------------------------------------------
__global__ __launch_bounds__(256) void gemm_bf16_nt(
    const unsigned short* __restrict__ A, const unsigned short* __restrict__ Bm,
    float* __restrict__ Cm, int M, int N, int K) {
  __shared__ unsigned short sA[2][128 * 40];
  __shared__ unsigned short sB[2][128 * 40];
  const int bm = blockIdx.y * 128, bn = blockIdx.x * 128;
  const int tid = threadIdx.x, lane = tid & 31, wid = tid >> 5;
  const int wm = (wid >> 2) * 64, wn = (wid & 3) * 32;
  const int lr = lane & 15, lh = lane >> 4;

  const v8f vzero = {0.f, 0.f, 0.f, 0.f, 0.f, 0.f, 0.f, 0.f};
  v8f acc[4][2];
#pragma unroll
  for (int mi = 0; mi < 4; ++mi)
#pragma unroll
    for (int ni = 0; ni < 2; ++ni) acc[mi][ni] = vzero;

  // This thread stages two 16B chunks per tile: chunk c = tid*2 + j.
  const int c0 = tid * 2, r0 = c0 >> 2, cc0 = (c0 & 3) * 8;
  const int c1 = tid * 2 + 1, r1 = c1 >> 2, cc1 = (c1 & 3) * 8;

  auto issue = [&](int k0, int buf) {
    async_copy_b128(A + (size_t)(bm + r0) * K + k0 + cc0, &sA[buf][r0 * 40 + cc0]);
    async_copy_b128(A + (size_t)(bm + r1) * K + k0 + cc1, &sA[buf][r1 * 40 + cc1]);
    async_copy_b128(Bm + (size_t)(bn + r0) * K + k0 + cc0, &sB[buf][r0 * 40 + cc0]);
    async_copy_b128(Bm + (size_t)(bn + r1) * K + k0 + cc1, &sB[buf][r1 * 40 + cc1]);
  };

  issue(0, 0);
  wait_async0();
  __syncthreads();

  for (int k0 = 0; k0 < K; k0 += 32) {
    const int buf = (k0 >> 5) & 1;
    if (k0 + 32 < K) issue(k0 + 32, buf ^ 1);   // overlap copy with compute

    const unsigned short* bp0 = &sB[buf][(wn + lr) * 40 + lh * 8];
    const unsigned short* bp1 = &sB[buf][(wn + 16 + lr) * 40 + lh * 8];
    v16bf bf0 = load2(bp0, bp0 + 16);
    v16bf bf1 = load2(bp1, bp1 + 16);
#pragma unroll
    for (int mi = 0; mi < 4; ++mi) {
      const unsigned short* ap = &sA[buf][(wm + mi * 16 + lr) * 40 + lh * 8];
      v16bf af = load2(ap, ap + 16);  // K {0..7,16..23} / {8..15,24..31}
      acc[mi][0] = wmma_bf16(af, bf0, acc[mi][0]);
      acc[mi][1] = wmma_bf16(af, bf1, acc[mi][1]);
    }
    wait_async0();
    __syncthreads();
  }
#pragma unroll
  for (int mi = 0; mi < 4; ++mi)
#pragma unroll
    for (int ni = 0; ni < 2; ++ni) {
      float* cp = Cm + (size_t)(bm + wm + mi * 16 + lh * 8) * N +
                  (bn + wn + ni * 16 + lr);
#pragma unroll
      for (int v = 0; v < 8; ++v) cp[(size_t)v * N] = acc[mi][ni][v];
    }
}

// ---------------------------------------------------------------------------
// RoPE on q,k + split qkv (fp32 [B,T,3C]) into bf16 q/k/v [B,H,T,D]
// ---------------------------------------------------------------------------
__global__ __launch_bounds__(256) void rope_split_kernel(
    const float* __restrict__ qkv, const float* __restrict__ rc,
    const float* __restrict__ rs, unsigned short* __restrict__ Q,
    unsigned short* __restrict__ Kq, unsigned short* __restrict__ V) {
  int idx = blockIdx.x * 256 + threadIdx.x;   // B*H*T*(D/2)
  int i = idx & 63;
  int t = (idx >> 6) & (Tt - 1);
  int h = (idx >> 17) & (Hh - 1);
  int b = idx >> 21;
  size_t base = ((size_t)(b * Tt + t)) * (3 * Cc) + h * Dd + 2 * i;
  float q0 = qkv[base], q1 = qkv[base + 1];
  float k0 = qkv[base + Cc], k1 = qkv[base + Cc + 1];
  float v0 = qkv[base + 2 * Cc], v1 = qkv[base + 2 * Cc + 1];
  float c = rc[t * (Dd / 2) + i], s = rs[t * (Dd / 2) + i];
  size_t o = (((size_t)(b * Hh + h) * Tt + t)) * Dd + 2 * i;
  Q[o] = f2bf(q0 * c - q1 * s);
  Q[o + 1] = f2bf(q0 * s + q1 * c);
  Kq[o] = f2bf(k0 * c - k1 * s);
  Kq[o + 1] = f2bf(k0 * s + k1 * c);
  V[o] = f2bf(v0);
  V[o + 1] = f2bf(v1);
}

// ---------------------------------------------------------------------------
// Flash attention: grid (T/64, B*H), 128 threads = 4 waves, each wave owns a
// 16-row query tile with full D=128. Key blocks of 32, online softmax.
// ---------------------------------------------------------------------------
__global__ __launch_bounds__(128) void attn_kernel(
    const unsigned short* __restrict__ Q, const unsigned short* __restrict__ Kq,
    const unsigned short* __restrict__ V, unsigned short* __restrict__ Y) {
  __shared__ unsigned short s_vt[4 * 128 * 40];  // per-wave V^T [dim][key]
  __shared__ unsigned short s_ps[4 * 16 * 32];   // per-wave P (16x32) bf16
  const int tid = threadIdx.x, lane = tid & 31, wid = tid >> 5;
  const int lr = lane & 15, lh = lane >> 4;
  const int bh = blockIdx.y;
  const int b = bh >> 4, h = bh & 15;
  const int m0 = blockIdx.x * 64 + wid * 16;
  unsigned short* vt = s_vt + wid * (128 * 40);
  unsigned short* ps = s_ps + wid * (16 * 32);
  const unsigned short* qbh = Q + (size_t)bh * Tt * Dd;
  const unsigned short* kbh = Kq + (size_t)bh * Tt * Dd;
  const unsigned short* vbh = V + (size_t)bh * Tt * Dd;

  // Q tile: 4 A-fragments (16x32 each) held in registers
  v16bf qf[4];
  {
    const unsigned short* qr = qbh + (size_t)(m0 + lr) * Dd;
#pragma unroll
    for (int kk = 0; kk < 4; ++kk) {
      const unsigned short* p = qr + kk * 32 + lh * 8;
      qf[kk] = load2(p, p + 16);
    }
  }
  const v8f vzero = {0.f, 0.f, 0.f, 0.f, 0.f, 0.f, 0.f, 0.f};
  v8f o[8];
  float mrun[8], lrun[8];
#pragma unroll
  for (int v = 0; v < 8; ++v) {
    o[v] = vzero;
    mrun[v] = -1e30f;
    lrun[v] = 0.0f;
  }
  const float scale = 0.08838834764831845f;  // 1/sqrt(128)

  for (int kb = 0; kb < m0 + 16; kb += 32) {
    // stage V^T in LDS: lane owns key kb+lane
    {
      const unsigned short* vr = vbh + (size_t)(kb + lane) * Dd;
#pragma unroll
      for (int d0 = 0; d0 < Dd; d0 += 8) {
        union { uint4 u; unsigned short s[8]; } x;
        x.u = *(const uint4*)(vr + d0);
#pragma unroll
        for (int e = 0; e < 8; ++e) vt[(d0 + e) * 40 + lane] = x.s[e];
      }
    }
    // S = Q K^T : two 16x16 tiles, K=128 in 4 chunks
    v8f st[2] = {vzero, vzero};
#pragma unroll
    for (int t = 0; t < 2; ++t) {
      const unsigned short* kr = kbh + (size_t)(kb + t * 16 + lr) * Dd + lh * 8;
#pragma unroll
      for (int kk = 0; kk < 4; ++kk) {
        v16bf kf = load2(kr + kk * 32, kr + kk * 32 + 16);
        st[t] = wmma_bf16(qf[kk], kf, st[t]);
      }
    }
    // online softmax per row (row = m0 + v + 8*lh, 16 cols live in a half-wave)
    float alpha[8];
#pragma unroll
    for (int v = 0; v < 8; ++v) {
      int row = m0 + v + lh * 8;
      float x0 = (kb + lr <= row) ? st[0][v] * scale : -1e30f;
      float x1 = (kb + 16 + lr <= row) ? st[1][v] * scale : -1e30f;
      float mx = fmaxf(x0, x1);
#pragma unroll
      for (int off = 1; off < 16; off <<= 1) mx = fmaxf(mx, __shfl_xor(mx, off, 32));
      float mnew = fmaxf(mrun[v], mx);
      float a = __expf(mrun[v] - mnew);
      float p0 = __expf(x0 - mnew), p1 = __expf(x1 - mnew);
      float rsum = p0 + p1;
#pragma unroll
      for (int off = 1; off < 16; off <<= 1) rsum += __shfl_xor(rsum, off, 32);
      lrun[v] = lrun[v] * a + rsum;
      mrun[v] = mnew;
      alpha[v] = a;
      int pr = (v + lh * 8) * 32;  // re-layout P via LDS into A-fragment order
      ps[pr + lr] = f2bf(p0);
      ps[pr + 16 + lr] = f2bf(p1);
    }
#pragma unroll
    for (int n = 0; n < 8; ++n)
#pragma unroll
      for (int v = 0; v < 8; ++v) o[n][v] *= alpha[v];
    // O += P (16x32) * V (32x128): 8 WMMAs over dim chunks
    const unsigned short* pp = ps + lr * 32 + lh * 8;
    v16bf pf = load2(pp, pp + 16);
#pragma unroll
    for (int n = 0; n < 8; ++n) {
      const unsigned short* vp = vt + (n * 16 + lr) * 40 + lh * 8;
      v16bf vf = load2(vp, vp + 16);
      o[n] = wmma_bf16(pf, vf, o[n]);
    }
  }
  // write y (bf16) at [b, t, h*D + dim]
#pragma unroll
  for (int v = 0; v < 8; ++v) {
    float inv = 1.0f / lrun[v];
    int trow = m0 + v + lh * 8;
    unsigned short* yr = Y + ((size_t)(b * Tt + trow)) * Cc + h * Dd + lr;
#pragma unroll
    for (int n = 0; n < 8; ++n) yr[n * 16] = f2bf(o[n][v] * inv);
  }
}

// ---------------------------------------------------------------------------
extern "C" void kernel_launch(void* const* d_in, const int* in_sizes, int n_in,
                              void* d_out, int out_size, void* d_ws, size_t ws_size,
                              hipStream_t stream) {
  (void)in_sizes; (void)n_in; (void)out_size; (void)ws_size;
  const float* x = (const float*)d_in[0];
  const float* w_qkv = (const float*)d_in[1];
  const float* w_o = (const float*)d_in[2];
  const float* rope_cos = (const float*)d_in[3];
  const float* rope_sin = (const float*)d_in[4];
  float* out = (float*)d_out;

  char* ws = (char*)d_ws;
  size_t off = 0;
  auto alloc = [&](size_t bytes) -> char* {
    char* p = ws + off;
    off += (bytes + 255) & ~(size_t)255;
    return p;
  };
  const size_t NX = (size_t)Bb * Tt * Cc;       // 8388608
  const size_t NWQKV = (size_t)3 * Cc * Cc;     // 12582912
  const size_t NWO = (size_t)Cc * Cc;           // 4194304
  const size_t NQKV = (size_t)Bb * Tt * 3 * Cc; // 25165824

  unsigned short* xb = (unsigned short*)alloc(NX * 2);
  unsigned short* wqkvb = (unsigned short*)alloc(NWQKV * 2);
  unsigned short* wob = (unsigned short*)alloc(NWO * 2);
  float* qkvf = (float*)alloc(NQKV * 4);
  unsigned short* qb = (unsigned short*)alloc(NX * 2);
  unsigned short* kb = (unsigned short*)alloc(NX * 2);
  unsigned short* vb = (unsigned short*)alloc(NX * 2);
  unsigned short* yb = (unsigned short*)qkvf;   // qkv dead after rope: reuse

  cvt_bf16<<<(unsigned)((NX + 255) / 256), 256, 0, stream>>>(x, xb, (int)NX);
  cvt_bf16<<<(unsigned)((NWQKV + 255) / 256), 256, 0, stream>>>(w_qkv, wqkvb, (int)NWQKV);
  cvt_bf16<<<(unsigned)((NWO + 255) / 256), 256, 0, stream>>>(w_o, wob, (int)NWO);

  // qkv = x @ w_qkv^T   (M=4096, N=6144, K=2048)
  gemm_bf16_nt<<<dim3(6144 / 128, 4096 / 128), 256, 0, stream>>>(
      xb, wqkvb, qkvf, Bb * Tt, 3 * Cc, Cc);

  rope_split_kernel<<<(Bb * Hh * Tt * (Dd / 2)) / 256, 256, 0, stream>>>(
      qkvf, rope_cos, rope_sin, qb, kb, vb);

  attn_kernel<<<dim3(Tt / 64, Bb * Hh), 128, 0, stream>>>(qb, kb, vb, yb);

  // out = y @ w_o^T   (M=4096, N=2048, K=2048)
  gemm_bf16_nt<<<dim3(2048 / 128, 4096 / 128), 256, 0, stream>>>(
      yb, wob, out, Bb * Tt, Cc, Cc);
}
